// GCN_Encoder_68049461838526
// MI455X (gfx1250) — matrix operand order repaired and checked by reading, not compiled, not run on hardware
//
#include <hip/hip_runtime.h>
#include <hip/hip_bf16.h>
#include <stdint.h>

typedef __attribute__((ext_vector_type(16))) __bf16 v16bf;
typedef __attribute__((ext_vector_type(8)))  float  v8f;
typedef unsigned short ushort_t;

static __device__ __forceinline__ unsigned short f2bf_rne(float f) {
    unsigned int u = __float_as_uint(f);
    unsigned int r = u + 0x7FFFu + ((u >> 16) & 1u);
    return (unsigned short)(r >> 16);
}
static __device__ __forceinline__ float bf2f(unsigned short h) {
    return __uint_as_float(((unsigned int)h) << 16);
}

// gfx1250 async global->LDS copy (ASYNCcnt path), 16 bytes per lane.
static __device__ __forceinline__ void async_copy_b128(unsigned lds_off,
                                                       unsigned long long gaddr) {
    asm volatile("global_load_async_to_lds_b128 %0, %1, off"
                 :: "v"(lds_off), "v"(gaddr) : "memory");
}
static __device__ __forceinline__ void wait_async0() {
    asm volatile("s_wait_asynccnt 0x0" ::: "memory");
}

// ---------------------------------------------------------------- utilities
__global__ void zero_f32_kernel(float* p, int n) {
    int i = blockIdx.x * blockDim.x + threadIdx.x;
    if (i < n) p[i] = 0.f;
}

__global__ void deg_accum_kernel(const long long* __restrict__ dst, float* __restrict__ deg, int E) {
    int e = blockIdx.x * blockDim.x + threadIdx.x;
    if (e < E) atomicAdd(&deg[dst[e]], 1.0f);
}

__global__ void dinv_kernel(float* deg, int n) {  // deg -> rsqrt(deg+1) in place
    int i = blockIdx.x * blockDim.x + threadIdx.x;
    if (i < n) deg[i] = rsqrtf(deg[i] + 1.0f);
}

// W[K x Nc] row-major  ->  Bt_hi/lo[Nc x Kpad] (transposed, zero-padded, bf16 hi/lo split)
__global__ void transpose_split_kernel(const float* __restrict__ W,
                                       ushort_t* __restrict__ hi, ushort_t* __restrict__ lo,
                                       int K, int Kpad, int Nc) {
    int i = blockIdx.x * blockDim.x + threadIdx.x;
    int total = Nc * Kpad;
    if (i < total) {
        int n = i / Kpad;
        int k = i % Kpad;
        float x = (k < K) ? W[(long long)k * Nc + n] : 0.f;
        unsigned short h = f2bf_rne(x);
        hi[i] = h;
        lo[i] = f2bf_rne(x - bf2f(h));
    }
}

// ------------------------------------------------------- bf16x3 WMMA GEMM
// C[M,Nc] = A[M,K] @ B[K,Nc] (+bias)(+relu).
// A fp32 (split to bf16 hi/lo on the fly); B pre-transposed/padded bf16 hi/lo [Nc x Kpad].
// Block = 256 threads (8 waves), 128x128 tile, K chunked by 32, ping-pong LDS,
// B tiles streamed with global_load_async_to_lds_b128.
__global__ __launch_bounds__(256)
void gemm_bf16x3_kernel(const float* __restrict__ A, int M, int K, int Kpad, int lda,
                        const ushort_t* __restrict__ Bhi, const ushort_t* __restrict__ Blo,
                        int Nc, float* __restrict__ C, int ldc,
                        const float* __restrict__ bias, int do_relu) {
    __shared__ __align__(16) ushort_t sAhi[2][128 * 32];
    __shared__ __align__(16) ushort_t sAlo[2][128 * 32];
    __shared__ __align__(16) ushort_t sBhi[2][128 * 32];   // [n][k]
    __shared__ __align__(16) ushort_t sBlo[2][128 * 32];

    const int t        = threadIdx.x;
    const int wave     = t >> 5;
    const int lane     = t & 31;
    const int m16      = lane & 15;
    const int hh       = lane >> 4;
    const int rowBase  = wave * 16;
    const int blockRow = blockIdx.x * 128;
    const int colTile  = blockIdx.y * 128;

    float av[16];

    // async-stream one B tile (hi+lo, 8KB each) into buffer `buf`
    auto issueB = [&](int buf, int kc) {
#pragma unroll
        for (int i = 0; i < 2; ++i) {
            int c  = i * 256 + t;      // 512 x 16B chunks per array
            int n  = c >> 2;
            int k0 = (c & 3) * 8;
            size_t gi = (size_t)(colTile + n) * Kpad + kc + k0;
            async_copy_b128((unsigned)(uintptr_t)&sBhi[buf][n * 32 + k0],
                            (unsigned long long)(uintptr_t)(Bhi + gi));
            async_copy_b128((unsigned)(uintptr_t)&sBlo[buf][n * 32 + k0],
                            (unsigned long long)(uintptr_t)(Blo + gi));
        }
    };
    // stage one A tile (128x32 fp32) into registers
    auto loadA = [&](int kc) {
#pragma unroll
        for (int i = 0; i < 16; ++i) {
            int e = i * 256 + t;
            int r = e >> 5, kk = e & 31;
            int gr = blockRow + r, gk = kc + kk;
            av[i] = (gr < M && gk < K) ? A[(long long)gr * lda + gk] : 0.f;
        }
    };
    // split + store staged A tile into buffer `buf`
    auto storeA = [&](int buf) {
#pragma unroll
        for (int i = 0; i < 16; ++i) {
            int e = i * 256 + t;
            unsigned short h = f2bf_rne(av[i]);
            sAhi[buf][e] = h;
            sAlo[buf][e] = f2bf_rne(av[i] - bf2f(h));
        }
    };

    v8f acc[8];
#pragma unroll
    for (int i = 0; i < 8; ++i) acc[i] = (v8f){0.f, 0.f, 0.f, 0.f, 0.f, 0.f, 0.f, 0.f};

    // prologue: fill buffer 0
    issueB(0, 0);
    loadA(0);
    storeA(0);
    wait_async0();
    __syncthreads();

    int cur = 0;
    for (int kc = 0; kc < Kpad; kc += 32) {
        const int  nxt     = cur ^ 1;
        const bool hasNext = (kc + 32) < Kpad;
        if (hasNext) {          // start next tile early: async B + global A loads in flight
            issueB(nxt, kc + 32);
            loadA(kc + 32);
        }

        // compute on `cur`
        union Frag { v16bf v; unsigned int ui[8]; } aHi, aLo;
        const int arow = (rowBase + m16) * 32;
#pragma unroll
        for (int v = 0; v < 8; ++v) {   // 16-bit A 16x32 layout (ISA 7.12.2)
            int k0 = (v < 4) ? (hh * 8 + 2 * v) : (16 + hh * 8 + 2 * (v - 4));
            aHi.ui[v] = *(const unsigned int*)&sAhi[cur][arow + k0];
            aLo.ui[v] = *(const unsigned int*)&sAlo[cur][arow + k0];
        }
#pragma unroll
        for (int nt = 0; nt < 8; ++nt) {
            union Frag bHi, bLo;
            const int brow = (nt * 16 + m16) * 32;
#pragma unroll
            for (int v = 0; v < 8; ++v) { // B 32x16: same K order as A (16-bit A 16x32 layout)
                int k0 = (v < 4) ? (hh * 8 + 2 * v) : (16 + hh * 8 + 2 * (v - 4));
                bHi.ui[v] = *(const unsigned int*)&sBhi[cur][brow + k0];
                bLo.ui[v] = *(const unsigned int*)&sBlo[cur][brow + k0];
            }
            acc[nt] = __builtin_amdgcn_wmma_f32_16x16x32_bf16(false, aHi.v, false, bHi.v,
                                                              (short)0, acc[nt], false, false);
            acc[nt] = __builtin_amdgcn_wmma_f32_16x16x32_bf16(false, aLo.v, false, bHi.v,
                                                              (short)0, acc[nt], false, false);
            acc[nt] = __builtin_amdgcn_wmma_f32_16x16x32_bf16(false, aHi.v, false, bLo.v,
                                                              (short)0, acc[nt], false, false);
        }

        if (hasNext) {
            storeA(nxt);
            wait_async0();
            __syncthreads();
            cur = nxt;
        }
    }

#pragma unroll
    for (int nt = 0; nt < 8; ++nt) {
        int col = colTile + nt * 16 + m16;
        float bv = bias ? bias[col] : 0.f;
#pragma unroll
        for (int v = 0; v < 8; ++v) {
            int row = blockRow + rowBase + hh * 8 + v;
            if (row < M) {
                float x = acc[nt][v] + bv;
                if (do_relu) x = fmaxf(x, 0.f);
                C[(long long)row * ldc + col] = x;
            }
        }
    }
}

// ----------------------------------------------------- GCN message passing
// agg initialized with self-loop term: agg[i,c] = hw[i,c]*dinv[i]^2 + bias[c]
__global__ void selfloop_init_kernel(const float* __restrict__ hw, const float* __restrict__ dinv,
                                     const float* __restrict__ bias, float* __restrict__ agg,
                                     int N, int H) {
    long long i = (long long)blockIdx.x * blockDim.x + threadIdx.x;
    long long total = (long long)N * H;
    if (i < total) {
        int node = (int)(i / H);
        int c = (int)(i % H);
        float d = dinv[node];
        agg[i] = hw[i] * d * d + bias[c];
    }
}

// one edge handled by H/4 threads, float4 gather + 4 atomic f32 adds
__global__ void scatter_edges_kernel(const long long* __restrict__ src, const long long* __restrict__ dst,
                                     const float* __restrict__ dinv, const float* __restrict__ hw,
                                     float* __restrict__ agg, int E, int H) {
    long long tid = (long long)blockIdx.x * blockDim.x + threadIdx.x;
    int lpe = H >> 2;
    long long total = (long long)E * lpe;
    if (tid >= total) return;
    int e  = (int)(tid / lpe);
    int c4 = (int)(tid % lpe) * 4;
    long long s = src[e], d = dst[e];
    float w = dinv[s] * dinv[d];
    const float4 v = *(const float4*)(hw + s * H + c4);
    float* o = agg + d * H + c4;
    atomicAdd(o + 0, v.x * w);
    atomicAdd(o + 1, v.y * w);
    atomicAdd(o + 2, v.z * w);
    atomicAdd(o + 3, v.w * w);
}

__global__ void relu_inplace_kernel(float* p, long long n) {
    long long i = (long long)blockIdx.x * blockDim.x + threadIdx.x;
    if (i < n) p[i] = fmaxf(p[i], 0.f);
}

// -------------------------------------------------------------- max pooling
__global__ void seg_starts_kernel(const long long* __restrict__ batch, int* __restrict__ starts,
                                  int N, int G) {
    int i = blockIdx.x * blockDim.x + threadIdx.x;
    if (i < N) {
        int b = (int)batch[i];
        if (i == 0 || (int)batch[i - 1] != b) starts[b] = i;
    }
    if (i == 0) starts[G] = N;
}

// one block per graph, one thread per feature
__global__ void pool_max_kernel(const float* __restrict__ h, const int* __restrict__ starts,
                                float* __restrict__ g, int H) {
    int gid = blockIdx.x;
    int c = threadIdx.x;
    int s = starts[gid], e = starts[gid + 1];
    float m = -3.402823466e38f;
    for (int r = s; r < e; ++r) m = fmaxf(m, h[(long long)r * H + c]);
    g[(long long)gid * H + c] = m;
}

// ------------------------------------------------------------------ driver
extern "C" void kernel_launch(void* const* d_in, const int* in_sizes, int n_in,
                              void* d_out, int out_size, void* d_ws, size_t ws_size,
                              hipStream_t stream) {
    const float*     x     = (const float*)d_in[0];
    const long long* ei    = (const long long*)d_in[1];
    const long long* batch = (const long long*)d_in[2];
    const float* W0  = (const float*)d_in[3];
    const float* b0  = (const float*)d_in[4];
    const float* W1  = (const float*)d_in[5];
    const float* b1  = (const float*)d_in[6];
    const float* Wf0 = (const float*)d_in[7];
    const float* bf0 = (const float*)d_in[8];
    const float* Wf1 = (const float*)d_in[9];
    const float* bf1 = (const float*)d_in[10];

    const int N    = in_sizes[2];
    const int E    = in_sizes[1] / 2;
    const int F_IN = in_sizes[0] / N;
    const int H    = in_sizes[4];      // 128
    const int FF0  = in_sizes[8];      // 256
    const int FF1  = in_sizes[10];     // 128
    const int G    = out_size / FF1;   // 4096

    const int Kp0  = (F_IN + 31) & ~31;   // 96
    const int Kp1  = (H + 31) & ~31;      // 128
    const int Kpf0 = (H + 31) & ~31;      // 128
    const int Kpf1 = (FF0 + 31) & ~31;    // 256

    const long long* src = ei;
    const long long* dst = ei + E;

    // workspace carve-out
    char* ws = (char*)d_ws;
    size_t off = 0;
    auto carve = [&](size_t bytes) -> char* {
        char* p = ws + off;
        off = (off + bytes + 255) & ~((size_t)255);
        return p;
    };
    float*    dinv   = (float*)carve((size_t)N * 4);
    float*    P0     = (float*)carve((size_t)N * H * 4);     // hw
    float*    P1     = (float*)carve((size_t)N * H * 4);     // agg / h
    float*    gpool  = (float*)carve((size_t)G * H * 4);
    float*    ff0    = (float*)carve((size_t)G * FF0 * 4);
    int*      starts = (int*)carve((size_t)(G + 1) * 4);
    // transposed / padded bf16 hi/lo weights: [Nc x Kpad]
    ushort_t* W0hi   = (ushort_t*)carve((size_t)H * Kp0 * 2);
    ushort_t* W0lo   = (ushort_t*)carve((size_t)H * Kp0 * 2);
    ushort_t* W1hi   = (ushort_t*)carve((size_t)H * Kp1 * 2);
    ushort_t* W1lo   = (ushort_t*)carve((size_t)H * Kp1 * 2);
    ushort_t* Wf0hi  = (ushort_t*)carve((size_t)FF0 * Kpf0 * 2);
    ushort_t* Wf0lo  = (ushort_t*)carve((size_t)FF0 * Kpf0 * 2);
    ushort_t* Wf1hi  = (ushort_t*)carve((size_t)FF1 * Kpf1 * 2);
    ushort_t* Wf1lo  = (ushort_t*)carve((size_t)FF1 * Kpf1 * 2);

    const int TB = 256;
    auto blk = [&](long long n) { return (unsigned)((n + TB - 1) / TB); };

    // degrees -> dinv
    zero_f32_kernel<<<blk(N), TB, 0, stream>>>(dinv, N);
    deg_accum_kernel<<<blk(E), TB, 0, stream>>>(dst, dinv, E);
    dinv_kernel<<<blk(N), TB, 0, stream>>>(dinv, N);

    // weight prep: transpose + pad + bf16 hi/lo split
    transpose_split_kernel<<<blk((long long)H * Kp0), TB, 0, stream>>>(W0, W0hi, W0lo, F_IN, Kp0, H);
    transpose_split_kernel<<<blk((long long)H * Kp1), TB, 0, stream>>>(W1, W1hi, W1lo, H, Kp1, H);
    transpose_split_kernel<<<blk((long long)FF0 * Kpf0), TB, 0, stream>>>(Wf0, Wf0hi, Wf0lo, H, Kpf0, FF0);
    transpose_split_kernel<<<blk((long long)FF1 * Kpf1), TB, 0, stream>>>(Wf1, Wf1hi, Wf1lo, FF0, Kpf1, FF1);

    const unsigned mtiles = (unsigned)((N + 127) / 128);
    const long long NH = (long long)N * H;
    const long long EW = (long long)E * (H / 4);

    // ---- GCN layer 0
    gemm_bf16x3_kernel<<<dim3(mtiles, H / 128), TB, 0, stream>>>(
        x, N, F_IN, Kp0, F_IN, W0hi, W0lo, H, P0, H, nullptr, 0);
    selfloop_init_kernel<<<blk(NH), TB, 0, stream>>>(P0, dinv, b0, P1, N, H);
    scatter_edges_kernel<<<blk(EW), TB, 0, stream>>>(src, dst, dinv, P0, P1, E, H);
    relu_inplace_kernel<<<blk(NH), TB, 0, stream>>>(P1, NH);

    // ---- GCN layer 1
    gemm_bf16x3_kernel<<<dim3(mtiles, H / 128), TB, 0, stream>>>(
        P1, N, H, Kp1, H, W1hi, W1lo, H, P0, H, nullptr, 0);
    selfloop_init_kernel<<<blk(NH), TB, 0, stream>>>(P0, dinv, b1, P1, N, H);
    scatter_edges_kernel<<<blk(EW), TB, 0, stream>>>(src, dst, dinv, P0, P1, E, H);
    relu_inplace_kernel<<<blk(NH), TB, 0, stream>>>(P1, NH);

    // ---- global max pool
    seg_starts_kernel<<<blk(N), TB, 0, stream>>>(batch, starts, N, G);
    pool_max_kernel<<<G, H, 0, stream>>>(P1, starts, gpool, H);

    // ---- FF head (bias + relu fused in GEMM epilogue)
    gemm_bf16x3_kernel<<<dim3((G + 127) / 128, FF0 / 128), TB, 0, stream>>>(
        gpool, G, H, Kpf0, H, Wf0hi, Wf0lo, FF0, ff0, FF0, bf0, 1);
    gemm_bf16x3_kernel<<<dim3((G + 127) / 128, FF1 / 128), TB, 0, stream>>>(
        ff0, G, FF0, Kpf1, FF0, Wf1hi, Wf1lo, FF1, (float*)d_out, FF1, bf1, 1);
}
